// DeformableMHA_30932354465799
// MI455X (gfx1250) — matrix operand order repaired and checked by reading, not compiled, not run on hardware
//
#include <hip/hip_runtime.h>
#include <hip/hip_bf16.h>
#include <math.h>

typedef __attribute__((ext_vector_type(16))) _Float16     v16h;
typedef __attribute__((ext_vector_type(8)))  _Float16     v8h;
typedef __attribute__((ext_vector_type(8)))  float        v8f;
typedef __attribute__((ext_vector_type(4)))  unsigned int u32x4;
typedef __attribute__((ext_vector_type(8)))  int          i32x8;
typedef __attribute__((ext_vector_type(4)))  int          i32x4;

#define N_CAMS 6
#define BB 2
#define KK 2048
#define DD 256
#define HH 128
#define WW 128
#define NH 8
#define NP 4
#define HD 32
#define ROWS (N_CAMS * BB * KK)   // 24576 query rows across cams
#define BK   (BB * KK)            // 4096
#define MAX_KD 1536               // largest GEMM K (final projection)

// ---------------------------------------------------------------------------
// Bilinear sample, align_corners=True, zero padding; exact gather semantics of
// the reference _grid_sample (per-corner validity mask, clamped index fetch).
// ---------------------------------------------------------------------------
__device__ __forceinline__ float bilin(const float* __restrict__ f, float ix, float iy) {
  float x0f = floorf(ix), y0f = floorf(iy);
  float wx1 = ix - x0f, wy1 = iy - y0f;
  float wx0 = 1.0f - wx1, wy0 = 1.0f - wy1;
  float acc = 0.0f;
#pragma unroll
  for (int dy = 0; dy < 2; ++dy) {
#pragma unroll
    for (int dx = 0; dx < 2; ++dx) {
      float xc = x0f + (float)dx, yc = y0f + (float)dy;
      bool v = (xc >= 0.0f) && (xc <= (float)(WW - 1)) && (yc >= 0.0f) && (yc <= (float)(HH - 1));
      int xi = (int)fminf(fmaxf(xc, 0.0f), (float)(WW - 1));
      int yi = (int)fminf(fmaxf(yc, 0.0f), (float)(HH - 1));
      float val = f[yi * WW + xi];
      float w = (dx ? wx1 : wx0) * (dy ? wy1 : wy0);
      acc += (v ? val : 0.0f) * w;
    }
  }
  return acc;
}

// ---------------------------------------------------------------------------
// Convert f32 row-major (Kd x N) weight to f16 column-major (N x Kd): one
// 16-column panel is then a contiguous Kd*16-element block, ideal for a single
// 1-D TDM tile per workgroup, and each lane's B fragment is contiguous 16B.
// ---------------------------------------------------------------------------
__global__ void convert_w_kernel(const float* __restrict__ W, _Float16* __restrict__ Wc,
                                 int Kd, int N) {
  int idx = blockIdx.x * blockDim.x + threadIdx.x;
  if (idx >= Kd * N) return;
  int k = idx % Kd, n = idx / Kd;
  Wc[idx] = (_Float16)W[(size_t)k * N + n];
}

// ---------------------------------------------------------------------------
// Stage 1: camera projection + first bilinear sample + residual add.
// One block per (cam, b, k); thread = channel c in [0,256).
// ---------------------------------------------------------------------------
__global__ void sample_cf_kernel(const float* __restrict__ feats,
                                 const float* __restrict__ cams,
                                 const float* __restrict__ points,
                                 const float* __restrict__ query,
                                 _Float16* __restrict__ Q,
                                 float* __restrict__ PIX) {
  int gq  = blockIdx.x;             // (cam*B + b)*K + k
  int cam = gq >> 12;               // /4096
  int rem = gq & 4095;
  int b   = rem >> 11;
  int k   = rem & 2047;
  int c   = threadIdx.x;

  const float* P  = points + ((size_t)b * KK + k) * 3;
  const float* Cm = cams + cam * 6;       // (3,2) row-major
  float p0 = P[0], p1 = P[1], p2 = P[2];
  float pix0 = (p0 * Cm[0] + p1 * Cm[2] + p2 * Cm[4]) * (float)HH;
  float pix1 = (p0 * Cm[1] + p1 * Cm[3] + p2 * Cm[5]) * (float)WW;
  // bound test on NORMALIZED coords (faithful to source quirk)
  float g0 = 2.0f * pix0 / (float)(HH - 1) - 1.0f;
  float g1 = 2.0f * pix1 / (float)(WW - 1) - 1.0f;
  float bound = (g0 >= 0.0f && g0 <= (float)HH && g1 >= 0.0f && g1 <= (float)WW) ? 1.0f : 0.0f;
  // with H==W the normalization round-trip cancels: ix = pix0, iy = pix1
  const float* fb = feats + (((size_t)cam * BB + b) * DD + c) * (HH * WW);
  float cf = bilin(fb, pix0, pix1);
  float qv = cf * bound + query[((size_t)b * KK + k) * DD + c];
  Q[(size_t)gq * DD + c] = (_Float16)qv;
  if (c == 0) { PIX[gq * 2 + 0] = pix0; PIX[gq * 2 + 1] = pix1; }
}

// ---------------------------------------------------------------------------
// WMMA f16 GEMM: C(MxN) = A(MxKd,f16,row-major) * B(KdxN via f16 col-major) + bias.
// Block = 8 waves = 128x16 output panel sharing one N-tile. The Kd x 16 B panel
// (contiguous in col-major) is staged ONCE per block into LDS by the Tensor
// Data Mover (tensor_load_to_lds, TENSORcnt), then read as ds_load_b128
// fragments. A streams from global with prefetch. Requires M % 128 == 0.
// xs_mode=1: write f16 into XS[(m%4096)*1536 + (m/4096)*256 + n] (fused concat).
// ---------------------------------------------------------------------------
__global__ void wmma_gemm_f16_kernel(const _Float16* __restrict__ A, int M, int Kd, int lda,
                                     const _Float16* __restrict__ Bc, int N,
                                     const float* __restrict__ bias,
                                     float* __restrict__ C, int ldc,
                                     _Float16* __restrict__ C16, int xs_mode) {
  __shared__ _Float16 sB[MAX_KD * 16];    // 48 KB of the WGP's 320 KB LDS

  int tilesN = N >> 4;
  int bm   = blockIdx.x / tilesN;
  int tn   = blockIdx.x % tilesN;
  int wave = threadIdx.x >> 5;
  int lane = threadIdx.x & 31;
  int tM   = (bm * 8 + wave) << 4;
  int tN   = tn << 4;
  int sub  = lane >> 4;                   // 0: lanes 0-15, 1: lanes 16-31
  int l    = lane & 15;

  // ---- TDM: stage the whole Kd x 16 B panel (contiguous Kd*16 halves) ------
  if (threadIdx.x < 32) {                 // wave 0 issues one DMA per block
    unsigned long long gaddr = (unsigned long long)(const void*)(Bc + (size_t)tN * Kd);
    unsigned int ldsaddr = (unsigned int)(uintptr_t)&sB[0];
    unsigned int nE = (unsigned int)(Kd * 16);     // elements (data_size = 2B), <= 24576
    u32x4 g0;
    g0[0] = 0x1u;                                  // count=1, user mode, no gather
    g0[1] = ldsaddr;                               // lds_addr [63:32]
    g0[2] = (unsigned int)(gaddr & 0xFFFFFFFFull); // global_addr [95:64]
    g0[3] = (unsigned int)((gaddr >> 32) & 0x01FFFFFFull) | (2u << 30); // addr hi + type=2
    i32x8 g1;
    g1[0] = (int)0x00010000u;                      // data_size = 1 (2 bytes)
    g1[1] = (int)((nE & 0xFFFFu) << 16);           // tensor_dim0 lo16
    g1[2] = (int)(((nE >> 16) & 0xFFFFu) | (1u << 16)); // tensor_dim0 hi16, tensor_dim1 = 1
    g1[3] = (int)((nE & 0xFFFFu) << 16);           // tensor_dim1 hi = 0, tile_dim0 = nE
    g1[4] = 0;                                     // tile_dim1/2 unused (1-D tile)
    g1[5] = (int)nE;                               // tensor_dim0_stride lo32
    g1[6] = 0;
    g1[7] = 0;
    i32x4 gz4 = {0, 0, 0, 0};
    i32x8 gz8 = {0, 0, 0, 0, 0, 0, 0, 0};
    // amdgpu-toolchain (clang-23) 6-arg form: (g0, g1, grp2, grp3, grp?, cpol)
    __builtin_amdgcn_tensor_load_to_lds(g0, g1, gz4, gz4, gz8, 0);
    __builtin_amdgcn_s_wait_tensorcnt(0);          // data resident in LDS
  }
  __syncthreads();

  // ---- WMMA main loop ------------------------------------------------------
  const _Float16* aRow = A + (size_t)(tM + l) * lda;
  int bBase = l * Kd;                     // lane's B column inside the LDS panel
  v8f acc = {};
  for (int k0 = 0; k0 < Kd; k0 += 32) {
    if (k0 + 32 < Kd) __builtin_prefetch((const void*)(aRow + k0 + 32), 0, 1);
    // A 16x32 f16 fragment (ISA 7.12.2): elems 0-7 = K k0+sub*8.., elems 8-15 = K k0+16+sub*8..
    v8h a0 = *(const v8h*)(aRow + k0 + sub * 8);
    v8h a1 = *(const v8h*)(aRow + k0 + 16 + sub * 8);
    // B 32x16 fragment from LDS: elems 0-7 = K k0+sub*8.., elems 8-15 = K k0+16+sub*8..
    v8h b0 = *(const v8h*)(&sB[bBase + k0 + sub * 8]);
    v8h b1 = *(const v8h*)(&sB[bBase + k0 + 16 + sub * 8]);
    v16h av, bv;
#pragma unroll
    for (int i = 0; i < 8; ++i) { av[i] = a0[i]; av[8 + i] = a1[i]; bv[i] = b0[i]; bv[8 + i] = b1[i]; }
    acc = __builtin_amdgcn_wmma_f32_16x16x32_f16(false, av, false, bv, (short)0, acc,
                                                 false, false);
  }

  // ---- Epilogue ------------------------------------------------------------
  int n = tN + l;
  float bvs = bias ? bias[n] : 0.0f;
#pragma unroll
  for (int i = 0; i < 8; ++i) {
    int m = tM + sub * 8 + i;             // C/D layout: VGPR i, lanes>=16 => M += 8
    float v = acc[i] + bvs;
    if (xs_mode) {
      // m = cam*4096 + (b*K + k); scatter into XS[(b,k)][cam*256 + n], ld = 1536
      C16[(size_t)(m & (BK - 1)) * (N_CAMS * DD) + ((m >> 12) << 8) + n] = (_Float16)v;
    } else {
      C[(size_t)m * ldc + n] = v;
    }
  }
}

// ---------------------------------------------------------------------------
// Stage 3: per-head softmax over 4 points + deformable bilinear gather.
// One block per (cam,b,k); thread t -> head = t/32, channel-in-head = t%32.
// ---------------------------------------------------------------------------
__global__ void sample_attend_kernel(const float* __restrict__ feats,
                                     const float* __restrict__ PIX,
                                     const float* __restrict__ OFF,
                                     const float* __restrict__ ATTL,
                                     _Float16* __restrict__ WF) {
  int gq  = blockIdx.x;
  int cam = gq >> 12;
  int rem = gq & 4095;
  int b   = rem >> 11;
  int t   = threadIdx.x;
  int h   = t >> 5;
  int d   = t & 31;

  float pix0 = PIX[gq * 2 + 0];
  float pix1 = PIX[gq * 2 + 1];

  const float* al = ATTL + (size_t)gq * (NH * NP) + h * NP;
  float l0 = al[0], l1 = al[1], l2 = al[2], l3 = al[3];
  float mx = fmaxf(fmaxf(l0, l1), fmaxf(l2, l3));
  float e0 = expf(l0 - mx), e1 = expf(l1 - mx), e2 = expf(l2 - mx), e3 = expf(l3 - mx);
  float inv = 1.0f / (e0 + e1 + e2 + e3);
  float w[4] = { e0 * inv, e1 * inv, e2 * inv, e3 * inv };

  const float* of = OFF + (size_t)gq * (NH * NP * 2) + h * (NP * 2);
  const float* fb = feats + (((size_t)cam * BB + b) * DD + h * HD + d) * (HH * WW);
  const float invH = 1.0f / (float)HH;    // off divided by hw = [128,128]
  float acc = 0.0f;
#pragma unroll
  for (int p = 0; p < NP; ++p) {
    float pc0 = pix0 + of[p * 2 + 0] * invH;
    float pc1 = pix1 + of[p * 2 + 1] * invH;
    acc += w[p] * bilin(fb, pc0, pc1);    // H==W: normalization cancels, ix=pc0, iy=pc1
  }
  WF[(size_t)gq * DD + h * HD + d] = (_Float16)acc;
}

// ---------------------------------------------------------------------------
extern "C" void kernel_launch(void* const* d_in, const int* in_sizes, int n_in,
                              void* d_out, int out_size, void* d_ws, size_t ws_size,
                              hipStream_t stream) {
  const float* query  = (const float*)d_in[0];
  const float* points = (const float*)d_in[1];
  const float* feats  = (const float*)d_in[2];
  const float* cams   = (const float*)d_in[3];
  const float* w_off  = (const float*)d_in[4];
  const float* b_off  = (const float*)d_in[5];
  const float* w_attn = (const float*)d_in[6];
  const float* b_attn = (const float*)d_in[7];
  const float* w_out  = (const float*)d_in[8];
  const float* b_out  = (const float*)d_in[9];
  const float* w_proj = (const float*)d_in[10];
  const float* b_proj = (const float*)d_in[11];
  float* out = (float*)d_out;

  // Workspace partition (256B aligned)
  char* base = (char*)d_ws;
  size_t off = 0;
  auto alloc = [&](size_t bytes) -> void* {
    void* p = base + off;
    off = (off + bytes + 255) & ~(size_t)255;
    return p;
  };
  _Float16* Q    = (_Float16*)alloc((size_t)ROWS * DD * sizeof(_Float16));
  float*    PIX  = (float*)   alloc((size_t)ROWS * 2 * sizeof(float));
  float*    OFF  = (float*)   alloc((size_t)ROWS * NH * NP * 2 * sizeof(float));
  float*    ATTL = (float*)   alloc((size_t)ROWS * NH * NP * sizeof(float));
  _Float16* WF   = (_Float16*)alloc((size_t)ROWS * DD * sizeof(_Float16));
  _Float16* XS   = (_Float16*)alloc((size_t)BK * N_CAMS * DD * sizeof(_Float16));
  _Float16* wc_off  = (_Float16*)alloc((size_t)DD * 64 * sizeof(_Float16));
  _Float16* wc_attn = (_Float16*)alloc((size_t)DD * 32 * sizeof(_Float16));
  _Float16* wc_out  = (_Float16*)alloc((size_t)DD * DD * sizeof(_Float16));
  _Float16* wc_proj = (_Float16*)alloc((size_t)(N_CAMS * DD) * DD * sizeof(_Float16));

  // 0) weight conversion to f16 column-major
  auto cvt = [&](const float* W, _Float16* Wc, int Kd, int N) {
    int n = Kd * N;
    convert_w_kernel<<<(n + 255) / 256, 256, 0, stream>>>(W, Wc, Kd, N);
  };
  cvt(w_off,  wc_off,  DD, 64);
  cvt(w_attn, wc_attn, DD, 32);
  cvt(w_out,  wc_out,  DD, DD);
  cvt(w_proj, wc_proj, N_CAMS * DD, DD);

  // 1) first-stage sample + residual
  sample_cf_kernel<<<ROWS, 256, 0, stream>>>(feats, cams, points, query, Q, PIX);

  // 2) offsets / attention-logit GEMMs (batched over all cams)
  //    grid = (M/128 row-panels) x (N/16 col-tiles); M always % 128 == 0 here.
  auto gemm_blocks = [](int M, int N) { return (M / 128) * (N / 16); };
  wmma_gemm_f16_kernel<<<gemm_blocks(ROWS, 64), 256, 0, stream>>>(
      Q, ROWS, DD, DD, wc_off, 64, b_off, OFF, 64, nullptr, 0);
  wmma_gemm_f16_kernel<<<gemm_blocks(ROWS, 32), 256, 0, stream>>>(
      Q, ROWS, DD, DD, wc_attn, 32, b_attn, ATTL, 32, nullptr, 0);

  // 3) deformable attention gather
  sample_attend_kernel<<<ROWS, 256, 0, stream>>>(feats, PIX, OFF, ATTL, WF);

  // 4) per-cam output projection, fused scatter into concatenated XS (f16)
  wmma_gemm_f16_kernel<<<gemm_blocks(ROWS, DD), 256, 0, stream>>>(
      WF, ROWS, DD, DD, wc_out, DD, b_out, nullptr, 0, XS, 1);

  // 5) final fused projection -> d_out (B*K x 256, f32)
  wmma_gemm_f16_kernel<<<gemm_blocks(BK, DD), 256, 0, stream>>>(
      XS, BK, N_CAMS * DD, N_CAMS * DD, wc_proj, DD, b_proj, out, DD, nullptr, 0);
}
